// KalmanNet_85426899517642
// MI455X (gfx1250) — matrix operand order repaired and checked by reading, not compiled, not run on hardware
//
#include <hip/hip_runtime.h>
#include <hip/hip_bf16.h>

// ---------------------------------------------------------------------------
// KalmanNet sequential scan, persistent single-workgroup kernel for gfx1250.
// All GEMVs on v_wmma_f32_16x16x32_f16 with prepacked f16 B-fragments in L2.
// Matvec input vectors kept as f16 in LDS (A-frag = 2x ds_load_b128), biases
// seeded unconditionally into WMMA accumulators, unrolling bounded to avoid
// scratch spills, small state math in wave-0 registers via wave32 __shfl.
// ---------------------------------------------------------------------------

typedef __attribute__((ext_vector_type(16))) _Float16 v16h;
typedef __attribute__((ext_vector_type(8)))  _Float16 v8h;
typedef __attribute__((ext_vector_type(8)))  float    v8f;

#define KN_N       16
#define KN_M       8
#define KN_TSTEPS  4096
#define KN_DT      0.01f
#define KN_NWAVES  16
#define KN_THREADS 512

// ---- fragment arena offsets (in halves). K padded to mult of 32, N to 16 ----
#define OFF_WXQ     0                         // 32  x 768
#define OFF_WHQ     (OFF_WXQ     + 32*768)    // 256 x 768
#define OFF_WSXXIN  (OFF_WHQ     + 256*768)   // 544 x 16
#define OFF_WSXX    (OFF_WSXXIN  + 544*16)    // 32  x 256
#define OFF_WY      (OFF_WSXX    + 32*256)    // 32  x 16
#define OFF_WSXXOUT (OFF_WY      + 32*16)     // 256 x 16
#define OFF_WXS     (OFF_WSXXOUT + 256*16)    // 32  x 192
#define OFF_WHS     (OFF_WXS     + 32*192)    // 64  x 192
#define OFF_WSXYIN  (OFF_WHS     + 64*192)    // 320 x 16
#define OFF_WSXY    (OFF_WSXYIN  + 320*16)    // 32  x 128
#define OFF_W1      (OFF_WSXY    + 32*128)    // 192 x 32
#define OFF_W2      (OFF_W1      + 192*32)    // 288 x 256
#define WS_HALVES   (OFF_W2      + 288*256)   // total = 350208 halves
#define CONST_BYTES ((size_t)WS_HALVES * 2)   // f32 constants A(256),B(16),H(128)

// ---------------------------------------------------------------------------
// Prep: pack fp32 weight (K x Nout, row-major) into WMMA B-fragment order:
// this code's contiguous-k B 32x16 (same k order as kn_build_a): lanes 0-15 hold K=0..15 of column n=lane, lanes 16-31 K=16..31
// ---------------------------------------------------------------------------
__global__ void kn_pack_w(const float* __restrict__ W, _Float16* __restrict__ Wf,
                          int K, int Nout, int Kc, int Ntiles)
{
    int total = Ntiles * Kc * 32 * 16;
    for (int idx = blockIdx.x * blockDim.x + threadIdx.x; idx < total;
         idx += gridDim.x * blockDim.x) {
        int i    = idx & 15;
        int lane = (idx >> 4) & 31;
        int q    = idx >> 9;
        int kc   = q % Kc;
        int nt   = q / Kc;
        int n    = nt * 16 + (lane & 15);
        int k    = kc * 32 + ((lane >> 4) << 4) + i;
        Wf[idx] = (k < K && n < Nout) ? (_Float16)W[(size_t)k * Nout + n]
                                      : (_Float16)0.0f;
    }
}

// deterministic stand-in for host-side numpy constants A_DYN / B_DYN / H_OBS
__device__ inline float kn_hrand(unsigned s)
{
    s ^= s * 747796405u + 2891336453u;
    s ^= s >> 17; s *= 0x9E3779B9u; s ^= s >> 13;
    return ((float)(s & 0xFFFFFFu) / 8388608.0f) - 1.0f;
}

__global__ void kn_gen_consts(float* __restrict__ A, float* __restrict__ Bc,
                              float* __restrict__ H)
{
    int t = blockIdx.x * blockDim.x + threadIdx.x;
    if (t < 256) {
        int i = t >> 4, j = t & 15;
        float mij = kn_hrand(1000u + (unsigned)(i * 16 + j));
        float mji = kn_hrand(1000u + (unsigned)(j * 16 + i));
        A[t] = 0.5f * (mij - mji) - (i == j ? 0.1f : 0.0f);
    }
    if (t < 16)  Bc[t] = 0.1f;
    if (t < 128) H[t] = 0.3f * kn_hrand(5000u + (unsigned)t);
}

// ---------------------------------------------------------------------------
// A-fragment from an f16 LDS vector: halves 0-7 = vin[kc*32+hb .. +7],
// halves 8-15 = vin[kc*32+hb+8 .. +7]  -> two ds_load_b128 + shuffle.
// ---------------------------------------------------------------------------
__device__ __forceinline__ v16h kn_build_a(const _Float16* __restrict__ vin,
                                           int kc, int hb)
{
    const _Float16* p = vin + (kc << 5) + hb;
    v8h lo = *(const v8h*)(p);
    v8h hi = *(const v8h*)(p + 8);
    return __builtin_shufflevector(lo, hi, 0, 1, 2, 3, 4, 5, 6, 7,
                                           8, 9, 10, 11, 12, 13, 14, 15);
}

// ---------------------------------------------------------------------------
// Wave-cooperative matvec: vout[0:NValid] = act(vin @ W + bias)
//  - bias seeded unconditionally into acc row M=0 (only row consumed)
//  - A fragments hoisted when a wave owns >1 tile; unroll bounded (no spills)
//  - OUT16: write f16 result (for buffers consumed only by other matvecs)
// ---------------------------------------------------------------------------
template<int KC, int NTILES, bool RELU, bool OUT16>
__device__ __forceinline__ void kn_matvec(const _Float16* __restrict__ Wf,
                                          const _Float16* __restrict__ vin,
                                          void* __restrict__ voutv,
                                          const float* __restrict__ biasLds,
                                          int NValid, int wave, int lane)
{
    if (NTILES < KN_NWAVES && wave >= NTILES) return;   // wave-uniform branch

    const int hb = (lane >> 4) << 4;   // A-fragment K sub-block (0 or 16)

    if constexpr (NTILES > KN_NWAVES) {
        // multiple tiles per wave: hoist all A fragments (KC<=8 here)
        v16h afrag[KC];
#pragma unroll
        for (int kc = 0; kc < KC; ++kc) afrag[kc] = kn_build_a(vin, kc, hb);
#pragma unroll 1
        for (int nt = wave; nt < NTILES; nt += KN_NWAVES) {
            v8f acc = {};
            acc[0] = biasLds[(nt << 4) + (lane & 15)];   // lanes>=16: row M=8, unused
            const _Float16* wp = Wf + (size_t)nt * KC * 512 + lane * 16;
#pragma unroll
            for (int kc = 0; kc < KC; ++kc) {
                const v16h b = *(const v16h*)(wp + kc * 512);
                acc = __builtin_amdgcn_wmma_f32_16x16x32_f16(
                          false, afrag[kc], false, b, (short)0, acc, false, false);
            }
            int n = (nt << 4) + lane;
            if (lane < 16 && n < NValid) {
                float r = RELU ? fmaxf(acc[0], 0.0f) : acc[0];
                if constexpr (OUT16) ((_Float16*)voutv)[n] = (_Float16)r;
                else                 ((float*)voutv)[n]    = r;
            }
        }
    } else {
        // single tile per wave; bounded unroll keeps register pressure low
        const int nt = wave;
        v8f acc = {};
        acc[0] = biasLds[(nt << 4) + (lane & 15)];
        const _Float16* wp = Wf + (size_t)nt * KC * 512 + lane * 16;
#pragma unroll 4
        for (int kc = 0; kc < KC; ++kc) {
            const v16h a = kn_build_a(vin, kc, hb);
            const v16h b = *(const v16h*)(wp + kc * 512);
            acc = __builtin_amdgcn_wmma_f32_16x16x32_f16(
                      false, a, false, b, (short)0, acc, false, false);
        }
        int n = (nt << 4) + lane;
        if (lane < 16 && n < NValid) {
            float r = RELU ? fmaxf(acc[0], 0.0f) : acc[0];
            if constexpr (OUT16) ((_Float16*)voutv)[n] = (_Float16)r;
            else                 ((float*)voutv)[n]    = r;
        }
    }
}

__device__ __forceinline__ float kn_sigmoid(float x)
{
    return 1.0f / (1.0f + __expf(-x));
}

struct KnSmem {
    // f32 state / VALU-consumed buffers
    float gQ[256], gSyy[64];
    float xm[768], hm[768], xmS[192], hmS[192];
    float Sxy[128], mII[64];
    // f16 matvec-input mirrors (padding stays zero after initial memset)
    alignas(16) _Float16 gQh[256];     // mirror of gQ
    alignas(16) _Float16 gSyyh[64];    // mirror of gSyy
    alignas(16) _Float16 Sxxh[256];    // Sxx (f16 only)
    alignas(16) _Float16 in544h[544];  // [Qv | dxt | gSxx(persistent, W2 out)]
    alignas(16) _Float16 in320h[320];  // [Sxx | invSyy]
    alignas(16) _Float16 in288h[288];  // [p1(W1 out) | Sxx | 0 0]
    alignas(16) _Float16 in192h[192];  // [invSyy | K]
    alignas(16) _Float16 in32xh[32], in32ah[32], in32bh[32], in32ch[32];
    // LDS-resident biases (padded to tile multiples)
    float bQ0[768], bQ1[768], bS0[192], bS1[192];
    float bbyL[16], bsxxinL[16], bsxxL[256], bsxxoutL[16];
    float bsxyinL[16], bsxyL[128], b1L[32], b2L[256];
};

__global__ void __launch_bounds__(KN_THREADS)
kn_persistent(const float* __restrict__ y_in,
              const _Float16* __restrict__ wf,
              const float* __restrict__ consts,
              const float* __restrict__ bQ,   const float* __restrict__ bS,
              const float* __restrict__ by,   const float* __restrict__ bsxxin,
              const float* __restrict__ bsxx, const float* __restrict__ bsxxout,
              const float* __restrict__ bsxyin, const float* __restrict__ bsxy,
              const float* __restrict__ b1,   const float* __restrict__ b2,
              float* __restrict__ out)
{
    __shared__ KnSmem sm;
    const int tid  = threadIdx.x;
    const int lane = tid & 31;
    const int wave = tid >> 5;

    { // zero all LDS (state init + persistent zero padding; 0 bits == f16 +0.0)
        unsigned* p = (unsigned*)&sm;
        for (int i = tid; i < (int)(sizeof(KnSmem) / 4); i += KN_THREADS) p[i] = 0u;
    }
    __syncthreads();

    // stage biases into LDS once
    for (int i = tid; i < 768; i += KN_THREADS) { sm.bQ0[i] = bQ[i]; sm.bQ1[i] = bQ[768 + i]; }
    for (int i = tid; i < 256; i += KN_THREADS) { sm.bsxxL[i] = bsxx[i]; sm.b2L[i] = b2[i]; }
    if (tid < 192) { sm.bS0[tid] = bS[tid]; sm.bS1[tid] = bS[192 + tid]; }
    if (tid < 128) sm.bsxyL[tid]   = bsxy[tid];
    if (tid < 30)  sm.b1L[tid]     = b1[tid];
    if (tid < 10)  sm.bbyL[tid]    = by[tid];
    if (tid < 10)  sm.bsxxinL[tid] = bsxxin[tid];
    if (tid < 10)  sm.bsxyinL[tid] = bsxyin[tid];
    if (tid < 5)   sm.bsxxoutL[tid] = bsxxout[tid];

    const float* Ad = consts;          // 16x16
    const float* Bd = consts + 256;    // 16
    const float* Hd = consts + 272;    // 8x16

    // wave-0 register-resident small state
    float Arow[16], Hrow[16];
    float Bdl = 0.0f, el = 0.0f, x1l = 0.0f, x2l = 0.0f, y1l = 0.0f, dyhl = 0.0f;
    if (wave == 0) {
#pragma unroll
        for (int j = 0; j < 16; ++j) Arow[j] = (lane < 16) ? Ad[lane * 16 + j] : 0.0f;
#pragma unroll
        for (int j = 0; j < 16; ++j) Hrow[j] = (lane < 8) ? Hd[lane * 16 + j] : 0.0f;
        Bdl = (lane < 16) ? Bd[lane] : 0.0f;
    }
    __syncthreads();

    const _Float16* fWxQ     = wf + OFF_WXQ;
    const _Float16* fWhQ     = wf + OFF_WHQ;
    const _Float16* fWsxxin  = wf + OFF_WSXXIN;
    const _Float16* fWsxx    = wf + OFF_WSXX;
    const _Float16* fWy      = wf + OFF_WY;
    const _Float16* fWsxxout = wf + OFF_WSXXOUT;
    const _Float16* fWxS     = wf + OFF_WXS;
    const _Float16* fWhS     = wf + OFF_WHS;
    const _Float16* fWsxyin  = wf + OFF_WSXYIN;
    const _Float16* fWsxy    = wf + OFF_WSXY;
    const _Float16* fW1      = wf + OFF_W1;
    const _Float16* fW2      = wf + OFF_W2;

    for (int t = 0; t < KN_TSTEPS; ++t) {
        // ---- stage A: RK4 + residuals, entirely in wave 0 (no barriers) ----
        if (wave == 0) {
            auto fode = [&](float x) -> float {
                float s = Bdl;
#pragma unroll
                for (int j = 0; j < 16; ++j) s += __shfl(x, j, 32) * Arow[j];
                return s;
            };
            float k1 = fode(el);
            float k2 = fode(el + 0.5f * KN_DT * k1);
            float k3 = fode(el + 0.5f * KN_DT * k2);
            float k4 = fode(el + KN_DT * k3);
            el += (KN_DT / 6.0f) * (k1 + 2.0f * k2 + 2.0f * k3 + k4);

            if (lane < 16) {
                sm.in32xh[lane]       = (_Float16)(x1l - el);    // del_x_hat
                sm.in544h[256 + lane] = (_Float16)(x1l - x2l);   // del_x_tilde
            }
            float ybl = (lane < 8) ? y_in[(size_t)t * KN_M + lane] : 0.0f;
            float ho = 0.0f;
#pragma unroll
            for (int j = 0; j < 16; ++j) ho += __shfl(el, j, 32) * Hrow[j];
            dyhl = ybl - ho;                                     // del_y_hat
            if (lane < 8) {
                sm.in32bh[lane]     = (_Float16)dyhl;
                sm.in32bh[8 + lane] = (_Float16)(ybl - y1l);     // del_y_tilde
                y1l = ybl;
            }
        }
        __syncthreads();

        // ---- GRU Q: xm = dxh@WxQ+b0 ; hm = gQ@WhQ+b1 ----
        kn_matvec<1, 48, false, false>(fWxQ, sm.in32xh, sm.xm, sm.bQ0, 768, wave, lane);
        kn_matvec<8, 48, false, false>(fWhQ, sm.gQh,    sm.hm, sm.bQ1, 768, wave, lane);
        __syncthreads();
        if (tid < 256) {
            float z  = kn_sigmoid(sm.xm[tid]       + sm.hm[tid]);
            float r  = kn_sigmoid(sm.xm[256 + tid] + sm.hm[256 + tid]);
            float hc = tanhf(sm.xm[512 + tid] + r * sm.hm[512 + tid]);
            float hn = z * sm.gQ[tid] + (1.0f - z) * hc;         // Q == new gQ
            sm.gQ[tid]  = hn;
            sm.gQh[tid] = (_Float16)hn;
        }
        __syncthreads();

        // Qv = (Q @ Q^T) straight into in544h[0:256]
        if (tid < 256) {
            int i = tid >> 4, j = tid & 15;
            float s = 0.0f;
#pragma unroll
            for (int k = 0; k < 16; ++k) s += sm.gQ[i * 16 + k] * sm.gQ[j * 16 + k];
            sm.in544h[tid] = (_Float16)s;
        }
        __syncthreads();

        // s10 = relu(in544 @ Wsxxin + b) -> in32a (slots 10..31 stay 0)
        kn_matvec<17, 1, true, true>(fWsxxin, sm.in544h, sm.in32ah, sm.bsxxinL, 10, wave, lane);
        __syncthreads();

        // Sxx = in32a @ Wsxx + b ;  py = relu(in32b @ Wy + b) -> in32c[0:10]
        kn_matvec<1, 16, false, true>(fWsxx, sm.in32ah, sm.Sxxh,  sm.bsxxL, 256, wave, lane);
        kn_matvec<1, 1,  true,  true>(fWy,   sm.in32bh, sm.in32ch, sm.bbyL, 10,  wave, lane);
        __syncthreads();

        // pSxx = relu(Sxx @ Wsxxout + b) -> in32c[10:15]
        kn_matvec<8, 1, true, true>(fWsxxout, sm.Sxxh, sm.in32ch + 10, sm.bsxxoutL, 5, wave, lane);
        __syncthreads();

        // ---- GRU S ----
        kn_matvec<1, 12, false, false>(fWxS, sm.in32ch, sm.xmS, sm.bS0, 192, wave, lane);
        kn_matvec<2, 12, false, false>(fWhS, sm.gSyyh,  sm.hmS, sm.bS1, 192, wave, lane);
        __syncthreads();
        if (tid < 64) {
            float z  = kn_sigmoid(sm.xmS[tid]      + sm.hmS[tid]);
            float r  = kn_sigmoid(sm.xmS[64 + tid] + sm.hmS[64 + tid]);
            float hc = tanhf(sm.xmS[128 + tid] + r * sm.hmS[128 + tid]);
            float hn = z * sm.gSyy[tid] + (1.0f - z) * hc;       // invSyy == new gSyy
            sm.gSyy[tid]         = hn;
            sm.gSyyh[tid]        = (_Float16)hn;
            sm.in320h[256 + tid] = (_Float16)hn;
            sm.in192h[tid]       = (_Float16)hn;
        }
        if (tid < 256) sm.in320h[tid] = sm.Sxxh[tid];
        __syncthreads();

        // pxy = relu(in320 @ Wsxyin + b) -> in32a
        kn_matvec<10, 1, true, true>(fWsxyin, sm.in320h, sm.in32ah, sm.bsxyinL, 10, wave, lane);
        __syncthreads();

        // Sxy = in32a @ Wsxy + b (f32, VALU consumer) ;  mII = mI @ mI^T
        kn_matvec<1, 8, false, false>(fWsxy, sm.in32ah, sm.Sxy, sm.bsxyL, 128, wave, lane);
        if (tid < 64) {
            int i = tid >> 3, j = tid & 7;
            float s = 0.0f;
#pragma unroll
            for (int k = 0; k < 8; ++k) s += sm.gSyy[i * 8 + k] * sm.gSyy[j * 8 + k];
            sm.mII[tid] = s;
        }
        __syncthreads();

        // K = mSxy @ mII -> in192h[64:192] ; stage Sxx into in288h[30:286]
        if (tid < 128) {
            int i = tid >> 3, j = tid & 7;
            float s = 0.0f;
#pragma unroll
            for (int k = 0; k < 8; ++k) s += sm.Sxy[i * 8 + k] * sm.mII[k * 8 + j];
            sm.in192h[64 + tid] = (_Float16)s;
        }
        if (tid < 256) sm.in288h[30 + tid] = sm.Sxxh[tid];
        __syncthreads();

        // p1 = relu(in192 @ W1 + b1) -> in288h[0:30]
        kn_matvec<6, 2, true, true>(fW1, sm.in192h, sm.in288h, sm.b1L, 30, wave, lane);
        __syncthreads();

        // gSxx' = relu(in288 @ W2 + b2) -> in544h[272:528] (persistent slot)
        // in parallel: wave 0 computes x1' = E + Kg @ dyh, writes output
        if (wave == 0) {
            float s = el;
#pragma unroll
            for (int j = 0; j < 8; ++j)
                s += (float)sm.in192h[64 + lane * 8 + j] * __shfl(dyhl, j, 32);
            if (lane < 16) out[(size_t)t * KN_N + lane] = s;
            x2l = x1l;
            x1l = s;        // lanes >=16 hold junk, never consumed
        }
        kn_matvec<9, 16, true, true>(fW2, sm.in288h, sm.in544h + 272, sm.b2L, 256, wave, lane);
        __syncthreads();
    }
}

// ---------------------------------------------------------------------------
extern "C" void kernel_launch(void* const* d_in, const int* in_sizes, int n_in,
                              void* d_out, int out_size, void* d_ws, size_t ws_size,
                              hipStream_t stream)
{
    (void)in_sizes; (void)n_in; (void)out_size; (void)ws_size;

    const float* y_in     = (const float*)d_in[0];
    const float* WxQ      = (const float*)d_in[1];
    const float* WhQ      = (const float*)d_in[2];
    const float* bQ       = (const float*)d_in[3];
    const float* WxS      = (const float*)d_in[4];
    const float* WhS      = (const float*)d_in[5];
    const float* bS       = (const float*)d_in[6];
    const float* Wy       = (const float*)d_in[7];
    const float* by       = (const float*)d_in[8];
    const float* Wsxxin   = (const float*)d_in[9];
    const float* bsxxin   = (const float*)d_in[10];
    const float* Wsxx     = (const float*)d_in[11];
    const float* bsxx     = (const float*)d_in[12];
    const float* Wsxxout  = (const float*)d_in[13];
    const float* bsxxout  = (const float*)d_in[14];
    const float* Wsxyin   = (const float*)d_in[15];
    const float* bsxyin   = (const float*)d_in[16];
    const float* Wsxy     = (const float*)d_in[17];
    const float* bsxy     = (const float*)d_in[18];
    const float* W1       = (const float*)d_in[19];
    const float* b1       = (const float*)d_in[20];
    const float* W2       = (const float*)d_in[21];
    const float* b2       = (const float*)d_in[22];

    _Float16* wf   = (_Float16*)d_ws;
    float* consts  = (float*)((char*)d_ws + CONST_BYTES);

    struct PackDesc { const float* W; int off, K, Nout, Kc, Ntiles; };
    const PackDesc packs[] = {
        { WxQ,     OFF_WXQ,      16, 768,  1, 48 },
        { WhQ,     OFF_WHQ,     256, 768,  8, 48 },
        { Wsxxin,  OFF_WSXXIN,  528,  10, 17,  1 },
        { Wsxx,    OFF_WSXX,     10, 256,  1, 16 },
        { Wy,      OFF_WY,       16,  10,  1,  1 },
        { Wsxxout, OFF_WSXXOUT, 256,   5,  8,  1 },
        { WxS,     OFF_WXS,      15, 192,  1, 12 },
        { WhS,     OFF_WHS,      64, 192,  2, 12 },
        { Wsxyin,  OFF_WSXYIN,  320,  10, 10,  1 },
        { Wsxy,    OFF_WSXY,     10, 128,  1,  8 },
        { W1,      OFF_W1,      192,  30,  6,  2 },
        { W2,      OFF_W2,      286, 256,  9, 16 },
    };
    for (const PackDesc& p : packs) {
        int total  = p.Ntiles * p.Kc * 32 * 16;
        int blocks = (total + 255) / 256;
        hipLaunchKernelGGL(kn_pack_w, dim3(blocks), dim3(256), 0, stream,
                           p.W, wf + p.off, p.K, p.Nout, p.Kc, p.Ntiles);
    }
    hipLaunchKernelGGL(kn_gen_consts, dim3(1), dim3(256), 0, stream,
                       consts, consts + 256, consts + 272);

    hipLaunchKernelGGL(kn_persistent, dim3(1), dim3(KN_THREADS), 0, stream,
                       y_in, wf, consts,
                       bQ, bS, by, bsxxin, bsxx, bsxxout, bsxyin, bsxy, b1, b2,
                       (float*)d_out);
}
